// TopKSoftmax_14267881357588
// MI455X (gfx1250) — matrix operand order repaired and checked by reading, not compiled, not run on hardware
//
#include <hip/hip_runtime.h>

// ---------------------------------------------------------------- constants
#define N_ROWS 2048
#define DDIM   2048
#define VOCAB  50257
#define KSEL   128

#define BM 128
#define BN 128
#define BK 32
#define LDS_STRIDE 40          // BK + 8 bf16 pad -> conflict-free ds_load_b128
#define NKTILES (DDIM / BK)    // 64

typedef __attribute__((ext_vector_type(16))) __bf16 v16bf;
typedef __attribute__((ext_vector_type(8)))  float  v8f;
typedef int v4i __attribute__((ext_vector_type(4)));

// address-space-qualified v4i pointers for the async-LDS builtin
typedef __attribute__((address_space(1))) v4i* gbl_v4i_p;   // global
typedef __attribute__((address_space(3))) v4i* lds_v4i_p;   // LDS

// ---- gfx1250 async global->LDS copy (guarded; falls back to sync copy) ----
#if defined(__HIP_DEVICE_COMPILE__) && __has_builtin(__builtin_amdgcn_global_load_async_to_lds_b128)
#define HAVE_ASYNC_LDS 1
#else
#define HAVE_ASYNC_LDS 0
#endif

__device__ __forceinline__ void copy16_g2l(const __bf16* g, __bf16* l) {
#if HAVE_ASYNC_LDS
    gbl_v4i_p gp = (gbl_v4i_p)(uintptr_t)g;
    lds_v4i_p lp = (lds_v4i_p)(unsigned)(uintptr_t)l;
    __builtin_amdgcn_global_load_async_to_lds_b128(gp, lp, 0, 0);
#else
    *reinterpret_cast<uint4*>(l) = *reinterpret_cast<const uint4*>(g);
#endif
}

__device__ __forceinline__ void wait_async_copies() {
#if HAVE_ASYNC_LDS
#if __has_builtin(__builtin_amdgcn_s_wait_asynccnt)
    __builtin_amdgcn_s_wait_asynccnt(0);
#else
    asm volatile("s_wait_asynccnt 0" ::: "memory");
#endif
#endif
}

// ---------------------------------------------------------------- kernel 1
// fp32 -> bf16 convert of the 2048x2048 activation matrix into workspace.
__global__ void __launch_bounds__(256) cvt_bf16_kernel(const float* __restrict__ in,
                                                       __bf16* __restrict__ outbf) {
    const size_t i = ((size_t)blockIdx.x * 256 + threadIdx.x) * 4;
    const float4 f = *reinterpret_cast<const float4*>(in + i);
    __bf16 t[4];
    t[0] = (__bf16)f.x; t[1] = (__bf16)f.y; t[2] = (__bf16)f.z; t[3] = (__bf16)f.w;
    *reinterpret_cast<uint2*>(outbf + i) = *reinterpret_cast<const uint2*>(t);
}

// ---------------------------------------------------------------- kernel 2
// logits[n, v] = dot(input[n, :], W[v, :])   (bf16 inputs, fp32 accumulate)
// WG = 256 threads = 8 waves (2x4); each wave owns a 64x32 C tile
// (4x2 v_wmma_f32_16x16x32_bf16).  Ping-pong LDS, one barrier per K-step.
__global__ void __launch_bounds__(256) gemm_wmma_kernel(const __bf16* __restrict__ Abf,
                                                        const float*  __restrict__ W,
                                                        float* __restrict__ out) {
    __shared__ __bf16 As[2][BM * LDS_STRIDE];   // 2 x 10 KB
    __shared__ __bf16 Bs[2][BN * LDS_STRIDE];   // 2 x 10 KB

    const int tid   = threadIdx.x;
    const int lane  = tid & 31;
    const int wave  = tid >> 5;    // 0..7
    const int waveM = wave >> 2;   // 0..1  -> 64 rows each
    const int waveN = wave & 3;    // 0..3  -> 32 cols each
    const int blockM = blockIdx.x * BM;   // x = M blocks: W tile shared by
    const int blockV = blockIdx.y * BN;   //     consecutive WGs -> L2 reuse

    v8f acc[4][2] = {};

    // staging assignment: thread t stages row t>>1, 16-element half t&1
    const int srow  = tid >> 1;    // 0..127
    const int shalf = tid & 1;     // 0..1
    const int vrow  = blockV + srow;
    const bool vok  = vrow < VOCAB;

    const int r = lane & 15;
    // ISA 7.12.2 operand K offsets per lane half:
    const int koffA = (lane < 16) ? 0 : 8;    // A: {0..7,16..23} / {8..15,24..31}
    const int koffB = (lane < 16) ? 0 : 8;    // B: {0..7,16..23} / {8..15,24..31}

    // ---- hoisted OOB handling: zero OOB W rows in both buffers, once
    if (!vok) {
        const uint4 z = make_uint4(0u, 0u, 0u, 0u);
#pragma unroll
        for (int b = 0; b < 2; ++b) {
            *reinterpret_cast<uint4*>(&Bs[b][srow * LDS_STRIDE + shalf * 16])     = z;
            *reinterpret_cast<uint4*>(&Bs[b][srow * LDS_STRIDE + shalf * 16 + 8]) = z;
        }
    }

    // ---- tile staging (A async to LDS; W sync with fp32->bf16 convert)
    auto stage = [&](int kk, int b) {
        const __bf16* ga = Abf + (size_t)(blockM + srow) * DDIM + kk + shalf * 16;
        __bf16* la = &As[b][srow * LDS_STRIDE + shalf * 16];
        copy16_g2l(ga, la);
        copy16_g2l(ga + 8, la + 8);

        if (vok) {
            const float4* wq = reinterpret_cast<const float4*>(
                W + (size_t)vrow * DDIM + kk + shalf * 16);
            __bf16 tmp[16];
#pragma unroll
            for (int c = 0; c < 4; ++c) {
                const float4 f = wq[c];
                tmp[c * 4 + 0] = (__bf16)f.x;
                tmp[c * 4 + 1] = (__bf16)f.y;
                tmp[c * 4 + 2] = (__bf16)f.z;
                tmp[c * 4 + 3] = (__bf16)f.w;
            }
            __builtin_prefetch(reinterpret_cast<const char*>(wq) + BK * 4, 0, 1);
            *reinterpret_cast<uint4*>(&Bs[b][srow * LDS_STRIDE + shalf * 16])     =
                *reinterpret_cast<const uint4*>(&tmp[0]);
            *reinterpret_cast<uint4*>(&Bs[b][srow * LDS_STRIDE + shalf * 16 + 8]) =
                *reinterpret_cast<const uint4*>(&tmp[8]);
        }
    };

    union Op { v16bf v; uint4 q[2]; };

    stage(0, 0);
    for (int t = 0; t < NKTILES; ++t) {
        const int b = t & 1;
        wait_async_copies();
        __syncthreads();                       // tile t visible to all waves
        if (t + 1 < NKTILES) stage((t + 1) * BK, b ^ 1);

        Op a4[4], b2[2];
#pragma unroll
        for (int i = 0; i < 4; ++i) {
            const __bf16* ap = &As[b][(waveM * 64 + i * 16 + r) * LDS_STRIDE];
            a4[i].q[0] = *reinterpret_cast<const uint4*>(ap + koffA);
            a4[i].q[1] = *reinterpret_cast<const uint4*>(ap + koffA + 16);
        }
#pragma unroll
        for (int j = 0; j < 2; ++j) {
            const __bf16* bp = &Bs[b][(waveN * 32 + j * 16 + r) * LDS_STRIDE];
            b2[j].q[0] = *reinterpret_cast<const uint4*>(bp + koffB);
            b2[j].q[1] = *reinterpret_cast<const uint4*>(bp + koffB + 16);
        }
#pragma unroll
        for (int i = 0; i < 4; ++i)
#pragma unroll
            for (int j = 0; j < 2; ++j)
                acc[i][j] = __builtin_amdgcn_wmma_f32_16x16x32_bf16(
                    false, a4[i].v, false, b2[j].v, (short)0, acc[i][j], false, false);
    }

    // ---- epilogue: C layout is N = lane&15, M = e + 8*(lane>=16)
    const int mhi = (lane >= 16) ? 8 : 0;
#pragma unroll
    for (int i = 0; i < 4; ++i) {
#pragma unroll
        for (int j = 0; j < 2; ++j) {
            const int gv = blockV + waveN * 32 + j * 16 + r;
            if (gv < VOCAB) {
#pragma unroll
                for (int e = 0; e < 8; ++e) {
                    const int gm = blockM + waveM * 64 + i * 16 + mhi + e;
                    out[(size_t)gm * VOCAB + gv] = acc[i][j][e];
                }
            }
        }
    }
}

// ---------------------------------------------------------------- kernel 3
// Per-row: radix-select the 128th-largest logit, then masked softmax in place.
__device__ __forceinline__ unsigned int fmap(float f) {
    const unsigned int u = __float_as_uint(f);
    return (u & 0x80000000u) ? ~u : (u | 0x80000000u);   // monotone order map
}

__global__ void __launch_bounds__(256) topk_softmax_kernel(float* __restrict__ out,
                                                           const int* __restrict__ target) {
    const int row = blockIdx.x;
    const int tid = threadIdx.x;
    float* p = out + (size_t)row * VOCAB;
    const int tgt = target[row];

    __shared__ unsigned int hist[256];
    __shared__ float fred[256];
    __shared__ unsigned int s_prefix;
    __shared__ unsigned int s_rem;
    __shared__ float s_max;

    if (tid == 0) { s_prefix = 0u; s_rem = KSEL; }

    float lmax = -3.402823466e38f;
    unsigned int prefix = 0;

    for (int l = 0; l < 4; ++l) {
        hist[tid] = 0;
        __syncthreads();
        const int digshift = 24 - 8 * l;
        for (int i = tid; i < VOCAB; i += 256) {
            const float f = p[i];
            if (l == 0) lmax = fmaxf(lmax, f);
            const unsigned int u = fmap(f);
            if (l == 0 || (u >> (32 - 8 * l)) == prefix)
                atomicAdd(&hist[(u >> digshift) & 0xFFu], 1u);
        }
        if (l == 0) fred[tid] = lmax;
        __syncthreads();
        if (l == 0) {  // uniform branch: deterministic tree max
            for (int s = 128; s > 0; s >>= 1) {
                if (tid < s) fred[tid] = fmaxf(fred[tid], fred[tid + s]);
                __syncthreads();
            }
            if (tid == 0) s_max = fred[0];
        }
        if (tid == 0) {  // walk digits from the top, find bucket of the k-th
            unsigned int rem = s_rem;
            int d = 255;
            for (; d > 0; --d) {
                const unsigned int c = hist[d];
                if (c >= rem) break;
                rem -= c;
            }
            s_prefix = (s_prefix << 8) | (unsigned int)d;
            s_rem = rem;
        }
        __syncthreads();
        prefix = s_prefix;
    }

    const unsigned int T = prefix;   // uint image of the k-th largest logit
    const float xmax = s_max;        // top-1 is always selected -> row max

    float ls = 0.0f;
    for (int i = tid; i < VOCAB; i += 256) {
        const float f = p[i];
        if (fmap(f) >= T || i == tgt) ls += __expf(f - xmax);
    }
    fred[tid] = ls;
    __syncthreads();
    for (int s = 128; s > 0; s >>= 1) {
        if (tid < s) fred[tid] += fred[tid + s];
        __syncthreads();
    }
    const float inv = 1.0f / fred[0];

    for (int i = tid; i < VOCAB; i += 256) {
        const float f = p[i];
        const bool sel = (fmap(f) >= T) || (i == tgt);
        p[i] = sel ? __expf(f - xmax) * inv : 0.0f;
    }
}

// ---------------------------------------------------------------- launcher
extern "C" void kernel_launch(void* const* d_in, const int* in_sizes, int n_in,
                              void* d_out, int out_size, void* d_ws, size_t ws_size,
                              hipStream_t stream) {
    const float* inp    = (const float*)d_in[0];   // [2048, 2048] fp32
    const float* W      = (const float*)d_in[1];   // [50257, 2048] fp32
    const int*   target = (const int*)  d_in[2];   // [2048] int32
    // d_in[3] = k (==128, baked in at compile time)
    float*  out = (float*)d_out;                   // [2048, 50257] fp32
    __bf16* Abf = (__bf16*)d_ws;                   // 8 MB bf16 activations

    // 1) activations fp32 -> bf16
    cvt_bf16_kernel<<<(N_ROWS * DDIM) / (256 * 4), 256, 0, stream>>>(inp, Abf);

    // 2) WMMA GEMM -> logits in d_out   (x = M fast => W-tile L2 reuse)
    dim3 grid(N_ROWS / BM, (VOCAB + BN - 1) / BN);
    gemm_wmma_kernel<<<grid, 256, 0, stream>>>(Abf, W, out);

    // 3) per-row top-k threshold + masked softmax, in place
    topk_softmax_kernel<<<N_ROWS, 256, 0, stream>>>(out, target);
}
